// MultiHeadAttention_3350074491182
// MI455X (gfx1250) — compile-verified, matrix operand order checked by reading, not run on hardware
//
#include <hip/hip_runtime.h>

// ---------------------------------------------------------------------------
// MI455X (gfx1250, wave32) multi-head attention forward.
//  * all GEMMs on v_wmma_f32_16x16x32_f16 (f32 accumulate)
//  * GEMMs: 32x64 per-wave tiles, ping-pong unroll-by-2 pipelining
//    (no inter-iteration register copies)
//  * attention: flash-style online softmax; K/V tiles double-buffered in LDS
//    via the Tensor Data Mover (tensor_load_to_lds + s_wait_tensorcnt),
//    overlapping the TDM fetch of block i+1 with compute on block i.
//    TDM pad_enable produces the padded ld=72 LDS layout directly.
// ---------------------------------------------------------------------------

#define BDIM 2
#define SDIM 2048
#define DDIM 768
#define HDIM 12
#define DK   64
#define NKB  (SDIM / 64)   // 32 key blocks

typedef _Float16 v16h __attribute__((ext_vector_type(16)));
typedef _Float16 v8h  __attribute__((ext_vector_type(8)));
typedef _Float16 v4h  __attribute__((ext_vector_type(4)));
typedef float    v8f  __attribute__((ext_vector_type(8)));

// ---------------------------------------------------------------------------
// Fragment loader for 16x32 (MxK) f16 operands, row-major, leading dim ld.
// ISA 7.12.2 (16-bit A 16x32): lanes 0-15: row M=lane, K={0..7,16..23};
// lanes 16-31: row M=lane-16, K={8..15,24..31}. B (32x16) is symmetric with
// N in place of M, so a B fragment loads from B^T row-major the same way.
// ---------------------------------------------------------------------------
__device__ __forceinline__ v16h load_frag_rm(const _Float16* p, int ld) {
  const int lane = threadIdx.x & 31;
  const int m    = lane & 15;
  const int ks   = (lane >> 4) << 3;          // 0 or 8
  const _Float16* r = p + (size_t)m * ld + ks;
  v8h lo = *(const v8h*)(r);                  // K = ks .. ks+7
  v8h hi = *(const v8h*)(r + 16);             // K = ks+16 .. ks+23
  return __builtin_shufflevector(lo, hi, 0,1,2,3,4,5,6,7,8,9,10,11,12,13,14,15);
}

__device__ __forceinline__ v8f wmma_f16(v16h a, v16h b, v8f c) {
  return __builtin_amdgcn_wmma_f32_16x16x32_f16(
      false, a, false, b, (short)0, c, false, false);
}

// ---------------------------------------------------------------------------
// Tensor Data Mover: 2D tile load (global -> LDS) with LDS padding
// (32 dwords data + 4 dwords pad per row => padded ld of 72 halves for a
// 64-half row). Descriptor packing per CDNA5 ISA §8.3/§8.4.
// ---------------------------------------------------------------------------
#if __has_builtin(__builtin_amdgcn_tensor_load_to_lds) && \
    __has_builtin(__builtin_amdgcn_s_wait_tensorcnt)
#define HAVE_TDM 1
typedef unsigned uint4v __attribute__((ext_vector_type(4)));
typedef int      int8v  __attribute__((ext_vector_type(8)));
typedef int      int4v  __attribute__((ext_vector_type(4)));

__device__ __forceinline__ unsigned lds_addr_u32(const void* p) {
  return (unsigned)(size_t)((__attribute__((address_space(3))) const void*)p);
}

__device__ __forceinline__ void tdm_load_tile_2d(
    unsigned lds_byte, const void* gptr,
    unsigned tensor_d0, unsigned tensor_d1, unsigned stride0,
    unsigned tile_d0, unsigned tile_d1) {
  size_t ga = (size_t)gptr;
  uint4v g0;
  g0.x = 1u;                                        // count=1, user mode
  g0.y = lds_byte;                                  // lds_addr
  g0.z = (unsigned)(ga & 0xffffffffu);              // global_addr[31:0]
  g0.w = (unsigned)((ga >> 32) & 0x01ffffffu) | (2u << 30);  // addr hi | type=2
  int8v g1;
  //  data_size=1 (2B)<<16 | pad_enable<<20 | pad_interval=4 (32dw)<<22 |
  //  pad_amount=3 (4dw)<<25 ; workgroup_mask=0 (not in a cluster)
  g1[0] = (int)((1u << 16) | (1u << 20) | (4u << 22) | (3u << 25));
  g1[1] = (int)(tensor_d0 << 16);                          // tensor_dim0[15:0]
  g1[2] = (int)((tensor_d0 >> 16) | (tensor_d1 << 16));    // dim0 hi | dim1 lo
  g1[3] = (int)((tensor_d1 >> 16) | (tile_d0 << 16));      // dim1 hi | tile0
  g1[4] = (int)tile_d1;                                    // tile1 | tile2=0
  g1[5] = (int)stride0;                                    // dim0_stride lo
  g1[6] = 0;
  g1[7] = 0;
  int4v z4 = {0, 0, 0, 0};
#if __clang_major__ >= 23
  int8v z8 = {0, 0, 0, 0, 0, 0, 0, 0};
  __builtin_amdgcn_tensor_load_to_lds(g0, g1, z4, z4, z8, 0);
#else
  __builtin_amdgcn_tensor_load_to_lds(g0, g1, z4, z4, 0);
#endif
}
#endif  // HAVE_TDM

// ---------------------------------------------------------------------------
// f32 -> f16 conversion
// ---------------------------------------------------------------------------
__global__ __launch_bounds__(256)
void cvt_f32_to_f16_kernel(const float* __restrict__ src,
                           _Float16* __restrict__ dst, int n4) {
  int i = blockIdx.x * blockDim.x + threadIdx.x;
  if (i < n4) {
    float4 v = *(const float4*)(src + (size_t)i * 4);
    v4h o = { (_Float16)v.x, (_Float16)v.y, (_Float16)v.z, (_Float16)v.w };
    *(v4h*)(dst + (size_t)i * 4) = o;
  }
}

// ---------------------------------------------------------------------------
// Projection GEMM: Y = x @ W^T + bias. 256 threads = 8 waves, block tile
// 128(M) x 128(N); each wave computes 32x64. Ping-pong unroll-by-2: even
// half computes from (a0,a1,b0[]) while (a0n,a1n,b1[]) prefetches, then the
// roles swap — no register copies between iterations.
// mode 0/1: store [B,H,S,DK]; mode 2: store V transposed [B,H,DK,S].
// ---------------------------------------------------------------------------
__global__ __launch_bounds__(256)
void proj_qkv_kernel(const _Float16* __restrict__ x,
                     const _Float16* __restrict__ w,
                     const float* __restrict__ bias,
                     _Float16* __restrict__ dst, int mode) {
  const int wave = threadIdx.x >> 5;
  const int lane = threadIdx.x & 31;
  const int mw = wave & 3, nw = wave >> 2;
  const int gm0 = blockIdx.x * 128 + mw * 32;
  const int gn0 = blockIdx.y * 128 + nw * 64;

  const _Float16* A0 = x + (size_t)gm0 * DDIM;
  const _Float16* A1 = A0 + (size_t)16 * DDIM;
  const _Float16* wr[4];
#pragma unroll
  for (int f = 0; f < 4; ++f) wr[f] = w + (size_t)(gn0 + f * 16) * DDIM;

  const v8f zero = {0.f,0.f,0.f,0.f,0.f,0.f,0.f,0.f};
  v8f c[2][4] = {{zero, zero, zero, zero}, {zero, zero, zero, zero}};

  v16h a0 = load_frag_rm(A0, DDIM);
  v16h a1 = load_frag_rm(A1, DDIM);
  v16h b0[4];
#pragma unroll
  for (int f = 0; f < 4; ++f) b0[f] = load_frag_rm(wr[f], DDIM);

  for (int k = 0; k < DDIM; k += 64) {
    // prefetch k+32 (always in range: DDIM is a multiple of 64)
    v16h a0n = load_frag_rm(A0 + k + 32, DDIM);
    v16h a1n = load_frag_rm(A1 + k + 32, DDIM);
    v16h b1[4];
#pragma unroll
    for (int f = 0; f < 4; ++f) b1[f] = load_frag_rm(wr[f] + k + 32, DDIM);

#pragma unroll
    for (int f = 0; f < 4; ++f) {
      c[0][f] = wmma_f16(a0, b0[f], c[0][f]);
      c[1][f] = wmma_f16(a1, b0[f], c[1][f]);
    }
    if (k + 64 < DDIM) {       // prefetch k+64 into the even-half registers
      a0 = load_frag_rm(A0 + k + 64, DDIM);
      a1 = load_frag_rm(A1 + k + 64, DDIM);
#pragma unroll
      for (int f = 0; f < 4; ++f) b0[f] = load_frag_rm(wr[f] + k + 64, DDIM);
    }
#pragma unroll
    for (int f = 0; f < 4; ++f) {
      c[0][f] = wmma_f16(a0n, b1[f], c[0][f]);
      c[1][f] = wmma_f16(a1n, b1[f], c[1][f]);
    }
  }

  const int n = lane & 15, hi = lane >> 4;
#pragma unroll
  for (int i = 0; i < 2; ++i)
#pragma unroll
    for (int f = 0; f < 4; ++f)
#pragma unroll
      for (int r = 0; r < 8; ++r) {
        int row = gm0 + i * 16 + r + hi * 8;   // [0, 4096)
        int col = gn0 + f * 16 + n;            // [0, 768)
        _Float16 hval = (_Float16)(c[i][f][r] + bias[col]);
        int bb_ = row >> 11, s = row & (SDIM - 1);
        int hh  = col >> 6,  d = col & (DK - 1);
        if (mode == 2)
          dst[(((size_t)(bb_ * HDIM + hh)) * DK + d) * SDIM + s] = hval;
        else
          dst[(((size_t)(bb_ * HDIM + hh)) * SDIM + s) * DK + d] = hval;
      }
}

// ---------------------------------------------------------------------------
// Flash attention: grid (S/128, H, B), 256 threads = 8 waves, 16 query rows
// per wave. K/V tiles double-buffered in LDS: TDM fetch of block i+1 overlaps
// compute on block i.
// ---------------------------------------------------------------------------
__global__ __launch_bounds__(256)
void attn_kernel(const _Float16* __restrict__ Qh,
                 const _Float16* __restrict__ Kh,
                 const _Float16* __restrict__ Vt,
                 _Float16* __restrict__ att) {
  __shared__ __align__(16) _Float16 ldsK[2][64][72];   // keys x d   (padded)
  __shared__ __align__(16) _Float16 ldsV[2][64][72];   // d    x key (padded)
  __shared__ __align__(16) _Float16 ldsP[8][16][72];   // per-wave P tiles

  const int wave = threadIdx.x >> 5;
  const int lane = threadIdx.x & 31;
  const int qb = blockIdx.x, h = blockIdx.y, b = blockIdx.z;
  const int head = b * HDIM + h;
  const int qs0  = qb * 128 + wave * 16;

  const _Float16* Qbase = Qh + ((size_t)head * SDIM + qs0) * DK;
  const _Float16* Khead = Kh + (size_t)head * SDIM * DK;
  const _Float16* Vhead = Vt + (size_t)head * DK * SDIM;
  _Float16* Pw = &ldsP[wave][0][0];

  v16h aq0 = load_frag_rm(Qbase, DK);
  v16h aq1 = load_frag_rm(Qbase + 32, DK);

  const v8f zero = {0.f,0.f,0.f,0.f,0.f,0.f,0.f,0.f};
  v8f of[4] = {zero, zero, zero, zero};
  float m_run[8], l_run[8];
#pragma unroll
  for (int r = 0; r < 8; ++r) { m_run[r] = -3.0e38f; l_run[r] = 0.f; }

  const int n = lane & 15, hi = lane >> 4;

#if defined(HAVE_TDM)
  const unsigned ldsKb[2] = { lds_addr_u32(&ldsK[0][0][0]),
                              lds_addr_u32(&ldsK[1][0][0]) };
  const unsigned ldsVb[2] = { lds_addr_u32(&ldsV[0][0][0]),
                              lds_addr_u32(&ldsV[1][0][0]) };
#endif

  // stage K/V tiles for key block kb into LDS buffer `buf` (no wait here)
  auto stage = [&](int buf, int kb) {
#if defined(HAVE_TDM)
    if (threadIdx.x < 32) {   // wave 0 drives the Tensor Data Mover
      tdm_load_tile_2d(ldsKb[buf], Khead + (size_t)kb * DK,
                       /*tensor_d0=*/DK, /*tensor_d1=*/SDIM, /*stride0=*/DK,
                       /*tile_d0=*/DK, /*tile_d1=*/64);
      tdm_load_tile_2d(ldsVb[buf], Vhead + kb,
                       /*tensor_d0=*/SDIM, /*tensor_d1=*/DK, /*stride0=*/SDIM,
                       /*tile_d0=*/64, /*tile_d1=*/DK);
    }
#else
    int row = threadIdx.x >> 2;            // 0..63
    int q   = (threadIdx.x & 3) * 16;      // 0,16,32,48
    const _Float16* ksrc = Khead + (size_t)(kb + row) * DK + q;
    const _Float16* vsrc = Vhead + (size_t)row * SDIM + kb + q;
    *(v8h*)&ldsK[buf][row][q]     = *(const v8h*)ksrc;
    *(v8h*)&ldsK[buf][row][q + 8] = *(const v8h*)(ksrc + 8);
    *(v8h*)&ldsV[buf][row][q]     = *(const v8h*)vsrc;
    *(v8h*)&ldsV[buf][row][q + 8] = *(const v8h*)(vsrc + 8);
#endif
  };

  stage(0, 0);   // prologue: buffer 0 <- key block 0

  for (int i = 0; i < NKB; ++i) {
    const int buf = i & 1;
#if defined(HAVE_TDM)
    if (threadIdx.x < 32) __builtin_amdgcn_s_wait_tensorcnt(0);
#endif
    // makes buf's tiles visible to all waves, and guarantees everyone is done
    // reading the buffer we are about to overwrite (read in the previous iteration)
    __syncthreads();
    if (i + 1 < NKB) stage(buf ^ 1, (i + 1) * 64);

    const _Float16 (*Kc)[72] = ldsK[buf];
    const _Float16 (*Vc)[72] = ldsV[buf];

    // ---- S = (Q K^T) * 1/sqrt(DK) -------------------------------------
    v8f sf[4];
#pragma unroll
    for (int f = 0; f < 4; ++f) {
      v16h b0 = load_frag_rm(&Kc[f * 16][0], 72);
      v16h b1 = load_frag_rm(&Kc[f * 16][0] + 32, 72);
      v8f s = zero;
      s = wmma_f16(aq0, b0, s);
      s = wmma_f16(aq1, b1, s);
      sf[f] = s * 0.125f;
    }

    // ---- online softmax (row reductions inside 16-lane halves) --------
    float rmax[8];
#pragma unroll
    for (int r = 0; r < 8; ++r)
      rmax[r] = fmaxf(fmaxf(sf[0][r], sf[1][r]), fmaxf(sf[2][r], sf[3][r]));
#pragma unroll
    for (int off = 1; off < 16; off <<= 1)
#pragma unroll
      for (int r = 0; r < 8; ++r)
        rmax[r] = fmaxf(rmax[r], __shfl_xor(rmax[r], off, 32));

    float alpha[8];
#pragma unroll
    for (int r = 0; r < 8; ++r) {
      float mn = fmaxf(m_run[r], rmax[r]);
      alpha[r] = __expf(m_run[r] - mn);
      m_run[r] = mn;
    }

    float rsum[8];
#pragma unroll
    for (int r = 0; r < 8; ++r) rsum[r] = 0.f;
#pragma unroll
    for (int f = 0; f < 4; ++f)
#pragma unroll
      for (int r = 0; r < 8; ++r) {
        float p = __expf(sf[f][r] - m_run[r]);
        sf[f][r] = p;
        rsum[r] += p;
      }
#pragma unroll
    for (int off = 1; off < 16; off <<= 1)
#pragma unroll
      for (int r = 0; r < 8; ++r)
        rsum[r] += __shfl_xor(rsum[r], off, 32);
#pragma unroll
    for (int r = 0; r < 8; ++r)
      l_run[r] = l_run[r] * alpha[r] + rsum[r];

#pragma unroll
    for (int f = 0; f < 4; ++f)
#pragma unroll
      for (int r = 0; r < 8; ++r)
        of[f][r] *= alpha[r];

    // ---- P: C-layout -> A-layout via LDS ------------------------------
    __syncthreads();
#pragma unroll
    for (int f = 0; f < 4; ++f)
#pragma unroll
      for (int r = 0; r < 8; ++r)
        Pw[(r + hi * 8) * 72 + f * 16 + n] = (_Float16)sf[f][r];
    __syncthreads();
    v16h pa0 = load_frag_rm(Pw, 72);
    v16h pa1 = load_frag_rm(Pw + 32, 72);

    // ---- O += P @ V ---------------------------------------------------
#pragma unroll
    for (int f = 0; f < 4; ++f) {
      v16h bv0 = load_frag_rm(&Vc[f * 16][0], 72);
      v16h bv1 = load_frag_rm(&Vc[f * 16][0] + 32, 72);
      of[f] = wmma_f16(pa0, bv0, of[f]);
      of[f] = wmma_f16(pa1, bv1, of[f]);
    }
  }

  // ---- normalize, store merged-head layout att[b*S+s, h*64+d] ---------
#pragma unroll
  for (int r = 0; r < 8; ++r) {
    float inv = 1.0f / l_run[r];
    int row = b * SDIM + qs0 + r + hi * 8;
#pragma unroll
    for (int f = 0; f < 4; ++f)
      att[(size_t)row * DDIM + h * DK + f * 16 + n] = (_Float16)(of[f][r] * inv);
  }
}

// ---------------------------------------------------------------------------
// Output projection: out = att @ wo^T + bo (fp32), same ping-pong pipeline.
// ---------------------------------------------------------------------------
__global__ __launch_bounds__(256)
void out_proj_kernel(const _Float16* __restrict__ att,
                     const _Float16* __restrict__ w,
                     const float* __restrict__ bias,
                     float* __restrict__ out) {
  const int wave = threadIdx.x >> 5;
  const int lane = threadIdx.x & 31;
  const int mw = wave & 3, nw = wave >> 2;
  const int gm0 = blockIdx.x * 128 + mw * 32;
  const int gn0 = blockIdx.y * 128 + nw * 64;

  const _Float16* A0 = att + (size_t)gm0 * DDIM;
  const _Float16* A1 = A0 + (size_t)16 * DDIM;
  const _Float16* wr[4];
#pragma unroll
  for (int f = 0; f < 4; ++f) wr[f] = w + (size_t)(gn0 + f * 16) * DDIM;

  const v8f zero = {0.f,0.f,0.f,0.f,0.f,0.f,0.f,0.f};
  v8f c[2][4] = {{zero, zero, zero, zero}, {zero, zero, zero, zero}};

  v16h a0 = load_frag_rm(A0, DDIM);
  v16h a1 = load_frag_rm(A1, DDIM);
  v16h b0[4];
#pragma unroll
  for (int f = 0; f < 4; ++f) b0[f] = load_frag_rm(wr[f], DDIM);

  for (int k = 0; k < DDIM; k += 64) {
    v16h a0n = load_frag_rm(A0 + k + 32, DDIM);
    v16h a1n = load_frag_rm(A1 + k + 32, DDIM);
    v16h b1[4];
#pragma unroll
    for (int f = 0; f < 4; ++f) b1[f] = load_frag_rm(wr[f] + k + 32, DDIM);

#pragma unroll
    for (int f = 0; f < 4; ++f) {
      c[0][f] = wmma_f16(a0, b0[f], c[0][f]);
      c[1][f] = wmma_f16(a1, b0[f], c[1][f]);
    }
    if (k + 64 < DDIM) {
      a0 = load_frag_rm(A0 + k + 64, DDIM);
      a1 = load_frag_rm(A1 + k + 64, DDIM);
#pragma unroll
      for (int f = 0; f < 4; ++f) b0[f] = load_frag_rm(wr[f] + k + 64, DDIM);
    }
#pragma unroll
    for (int f = 0; f < 4; ++f) {
      c[0][f] = wmma_f16(a0n, b1[f], c[0][f]);
      c[1][f] = wmma_f16(a1n, b1[f], c[1][f]);
    }
  }

  const int n = lane & 15, hi = lane >> 4;
#pragma unroll
  for (int i = 0; i < 2; ++i)
#pragma unroll
    for (int f = 0; f < 4; ++f)
#pragma unroll
      for (int r = 0; r < 8; ++r) {
        int row = gm0 + i * 16 + r + hi * 8;
        int col = gn0 + f * 16 + n;
        out[(size_t)row * DDIM + col] = c[i][f][r] + bias[col];
      }
}

// ---------------------------------------------------------------------------
// Host-side launcher
// ---------------------------------------------------------------------------
extern "C" void kernel_launch(void* const* d_in, const int* in_sizes, int n_in,
                              void* d_out, int out_size, void* d_ws, size_t ws_size,
                              hipStream_t stream) {
  (void)in_sizes; (void)n_in; (void)out_size; (void)ws_size;

  // setup_inputs() order: k, q, v, mask, wq, bq, wk, bk, wv, bv, wo, bo
  const float* in_k = (const float*)d_in[0];
  const float* in_q = (const float*)d_in[1];
  const float* in_v = (const float*)d_in[2];
  const float* wq   = (const float*)d_in[4];
  const float* bq   = (const float*)d_in[5];
  const float* wk   = (const float*)d_in[6];
  const float* bk   = (const float*)d_in[7];
  const float* wv   = (const float*)d_in[8];
  const float* bv   = (const float*)d_in[9];
  const float* wo   = (const float*)d_in[10];
  const float* bo   = (const float*)d_in[11];

  const size_t NX = (size_t)BDIM * SDIM * DDIM;   // 3,145,728
  const size_t NW = (size_t)DDIM * DDIM;          //   589,824

  char* ws = (char*)d_ws;
  size_t off = 0;
  auto take = [&](size_t elems) {
    _Float16* p = (_Float16*)(ws + off);
    off += (elems * sizeof(_Float16) + 255) & ~(size_t)255;
    return p;
  };
  _Float16* qh   = take(NX);
  _Float16* khx  = take(NX);
  _Float16* vhx  = take(NX);
  _Float16* wqh  = take(NW);
  _Float16* wkh  = take(NW);
  _Float16* wvh  = take(NW);
  _Float16* woh  = take(NW);
  _Float16* Qh   = take(NX);   // [B,H,S,DK]
  _Float16* Kh   = take(NX);   // [B,H,S,DK]
  _Float16* Vt   = take(NX);   // [B,H,DK,S]
  _Float16* atth = take(NX);   // [B*S, D]

  {
    int nx4 = (int)(NX / 4), nw4 = (int)(NW / 4);
    dim3 blk(256);
    dim3 gx((nx4 + 255) / 256), gw((nw4 + 255) / 256);
    cvt_f32_to_f16_kernel<<<gx, blk, 0, stream>>>(in_q, qh,  nx4);
    cvt_f32_to_f16_kernel<<<gx, blk, 0, stream>>>(in_k, khx, nx4);
    cvt_f32_to_f16_kernel<<<gx, blk, 0, stream>>>(in_v, vhx, nx4);
    cvt_f32_to_f16_kernel<<<gw, blk, 0, stream>>>(wq, wqh, nw4);
    cvt_f32_to_f16_kernel<<<gw, blk, 0, stream>>>(wk, wkh, nw4);
    cvt_f32_to_f16_kernel<<<gw, blk, 0, stream>>>(wv, wvh, nw4);
    cvt_f32_to_f16_kernel<<<gw, blk, 0, stream>>>(wo, woh, nw4);
  }

  {
    dim3 grid(BDIM * SDIM / 128, DDIM / 128);   // (32, 6)
    dim3 blk(256);
    proj_qkv_kernel<<<grid, blk, 0, stream>>>(qh,  wqh, bq, Qh, 0);
    proj_qkv_kernel<<<grid, blk, 0, stream>>>(khx, wkh, bk, Kh, 1);
    proj_qkv_kernel<<<grid, blk, 0, stream>>>(vhx, wvh, bv, Vt, 2);
  }

  attn_kernel<<<dim3(SDIM / 128, HDIM, BDIM), dim3(256), 0, stream>>>(Qh, Kh, Vt, atth);

  out_proj_kernel<<<dim3(BDIM * SDIM / 128, DDIM / 128), dim3(256), 0, stream>>>(
      atth, woh, bo, (float*)d_out);
}
